// ScatteringEncoder_82540681494613
// MI455X (gfx1250) — matrix operand order repaired and checked by reading, not compiled, not run on hardware
//
#include <hip/hip_runtime.h>
#include <hip/hip_bf16.h>

#define BATCH 8
#define ECH   384
#define HGT   64
#define WID   64
#define HEADS 8
#define DHEAD 48
#define TTOK  4096
#define HIDC  1536
#define EPSV  1e-5f

typedef __attribute__((ext_vector_type(16))) __bf16 v16bf;
typedef __attribute__((ext_vector_type(8)))  __bf16 v8bf;
typedef __attribute__((ext_vector_type(8)))  float  v8f;

__device__ __forceinline__ unsigned short f2bf(float f) {
  unsigned int u = __float_as_uint(f);
  unsigned int r = (u + 0x7FFFu + ((u >> 16) & 1u)) >> 16;  // RNE
  return (unsigned short)r;
}
__device__ __forceinline__ float bf2f(unsigned short s) {
  return __uint_as_float(((unsigned int)s) << 16);
}
__device__ __forceinline__ float mishf(float x) {
  float sp = (x > 20.f) ? x : log1pf(__expf(x));
  return x * tanhf(sp);
}

// ---- WMMA fragment loaders (bf16, 16x16x32) --------------------------------
// A (16x32, MxK): lane = M (0..15, both halves); VGPR0..3 -> K = hi*8 + e,
// VGPR4..7 -> K = 16 + hi*8 + (e-8)  => two contiguous 16B loads per lane.
__device__ __forceinline__ v16bf load_fragA(const unsigned short* row, int k0, int hi) {
  const v8bf* p0 = reinterpret_cast<const v8bf*>(row + k0 + hi * 8);
  const v8bf* p1 = reinterpret_cast<const v8bf*>(row + k0 + 16 + hi * 8);
  v8bf lo = *p0, hh = *p1;
  v16bf r;
#pragma unroll
  for (int i = 0; i < 8; ++i) { r[i] = lo[i]; r[i + 8] = hh[i]; }
  return r;
}
// B (32x16, KxN): lane = N; same K order as A (hi*8 + e, 16 + hi*8 + (e-8)) =>
// two contiguous 16B loads per lane from n-major rows.
__device__ __forceinline__ v16bf load_fragB(const unsigned short* row, int k0, int hi) {
  return load_fragA(row, k0, hi);
}

__device__ __forceinline__ v8f wmma_bf16(v16bf a, v16bf b, v8f c) {
  return __builtin_amdgcn_wmma_f32_16x16x32_bf16(false, a, false, b,
                                                 (short)0, c, false, false);
}

// ---- weight prep -----------------------------------------------------------
__global__ void k_prep_qkvw(const float* __restrict__ qw, const float* __restrict__ kw,
                            const float* __restrict__ vw,
                            const float* __restrict__ qg, const float* __restrict__ qvv,
                            const float* __restrict__ kg, const float* __restrict__ kvv,
                            const float* __restrict__ vg, const float* __restrict__ vvv,
                            unsigned short* __restrict__ wout) {
  int idx = blockIdx.x * 256 + threadIdx.x;
  if (idx >= 3 * ECH * ECH) return;
  int o = idx / ECH, i = idx % ECH;
  int s2 = o / ECH;       // 0=q,1=k,2=v
  int oo = o % ECH;
  float scale, wv;
  if (s2 == 0)      { scale = qg[oo] * rsqrtf(qvv[oo] + EPSV); wv = qw[oo * ECH + i]; }
  else if (s2 == 1) { scale = kg[oo] * rsqrtf(kvv[oo] + EPSV); wv = kw[oo * ECH + i]; }
  else              { scale = vg[oo] * rsqrtf(vvv[oo] + EPSV); wv = vw[oo * ECH + i]; }
  wout[idx] = f2bf(wv * scale);
}

__global__ void k_prep_qkvb(const float* qb, const float* qg, const float* qbe, const float* qm, const float* qvv,
                            const float* kb, const float* kg, const float* kbe, const float* km, const float* kvv,
                            const float* vb, const float* vg, const float* vbe, const float* vm, const float* vvv,
                            float* __restrict__ bout) {
  int idx = blockIdx.x * 256 + threadIdx.x;
  if (idx >= 3 * ECH) return;
  int sect = idx / ECH, oo = idx % ECH;
  const float *bp, *gp, *bep, *mp, *vp;
  if (sect == 0)      { bp = qb; gp = qg; bep = qbe; mp = qm; vp = qvv; }
  else if (sect == 1) { bp = kb; gp = kg; bep = kbe; mp = km; vp = kvv; }
  else                { bp = vb; gp = vg; bep = vbe; mp = vm; vp = vvv; }
  float s = gp[oo] * rsqrtf(vp[oo] + EPSV);
  bout[idx] = (bp[oo] - mp[oo]) * s + bep[oo];
}

__global__ void k_f2bfk(const float* __restrict__ src, unsigned short* __restrict__ dst, int n) {
  int i = blockIdx.x * 256 + threadIdx.x;
  if (i < n) dst[i] = f2bf(src[i]);
}

// ---- channel layernorm: (B,E,T) f32 -> (B,T,E) bf16 ------------------------
__global__ __launch_bounds__(256) void k_cln(const float* __restrict__ x,
                                             const float* __restrict__ w,
                                             const float* __restrict__ bb,
                                             unsigned short* __restrict__ outT) {
  int tid = threadIdx.x, wv = tid >> 5, lane = tid & 31;
  size_t pix = (size_t)blockIdx.x * 8 + wv;      // one wave per pixel
  int b = (int)(pix >> 12);
  int t = (int)(pix & (TTOK - 1));
  const float* base = x + (size_t)b * ECH * TTOK + t;
  float vals[12];
  float s = 0.f, sq = 0.f;
#pragma unroll
  for (int i = 0; i < 12; ++i) {
    int e = i * 32 + lane;
    float v = base[(size_t)e * TTOK];
    vals[i] = v; s += v; sq += v * v;
  }
  for (int o = 16; o > 0; o >>= 1) { s += __shfl_xor(s, o); sq += __shfl_xor(sq, o); }
  float mu = s * (1.f / ECH);
  float var = sq * (1.f / ECH) - mu * mu;
  float rs = rsqrtf(var + EPSV);
  unsigned short* orow = outT + pix * ECH;
#pragma unroll
  for (int i = 0; i < 12; ++i) {
    int e = i * 32 + lane;
    orow[e] = f2bf((vals[i] - mu) * rs * w[e] + bb[e]);
  }
}

// ---- bf16 WMMA GEMM: C[M,N] = A[M,K] * Bt[n-major] + bias ------------------
// 8 waves / block arranged 2(m) x 4(n); each wave owns a 32x32 register tile
// (2 A-frags x 2 B-frags -> 4 WMMAs per 32-wide K step).
// outF: f32 channel-major (+ optional addsrc); outT: bf16 n-major (chaining)
__global__ __launch_bounds__(256) void k_gemm(const unsigned short* __restrict__ A,
                                              const unsigned short* __restrict__ Bt,
                                              const float* __restrict__ bias,
                                              const float* __restrict__ addsrc,
                                              float* __restrict__ outF,
                                              unsigned short* __restrict__ outT,
                                              int M, int N, int K, int do_mish) {
  const int tid = threadIdx.x;
  const int wv = tid >> 5, lane = tid & 31, hi = lane >> 4, l15 = lane & 15;
  const int mBase = blockIdx.y * 64 + (wv >> 2) * 32;
  const int nBase = blockIdx.x * 128 + (wv & 3) * 32;
  const int bz = blockIdx.z;
  const unsigned short* arow0 = A + (size_t)(mBase + l15) * K;
  const unsigned short* arow1 = arow0 + (size_t)16 * K;
  const unsigned short* brow0 = Bt + (size_t)bz * N * K + (size_t)(nBase + l15) * K;
  const unsigned short* brow1 = brow0 + (size_t)16 * K;
  v8f acc00 = {0.f, 0.f, 0.f, 0.f, 0.f, 0.f, 0.f, 0.f};
  v8f acc01 = acc00, acc10 = acc00, acc11 = acc00;
  for (int k0 = 0; k0 < K; k0 += 32) {
    v16bf a0 = load_fragA(arow0, k0, hi);
    v16bf a1 = load_fragA(arow1, k0, hi);
    v16bf b0 = load_fragB(brow0, k0, hi);
    v16bf b1 = load_fragB(brow1, k0, hi);
    if (k0 + 32 < K) {
      __builtin_prefetch(arow0 + k0 + 32, 0, 3);   // global_prefetch_b8 (near)
      __builtin_prefetch(arow1 + k0 + 32, 0, 3);
      __builtin_prefetch(brow0 + k0 + 32, 0, 3);
      __builtin_prefetch(brow1 + k0 + 32, 0, 3);
    }
    acc00 = wmma_bf16(a0, b0, acc00);
    acc01 = wmma_bf16(a0, b1, acc01);
    acc10 = wmma_bf16(a1, b0, acc10);
    acc11 = wmma_bf16(a1, b1, acc11);
  }
#pragma unroll
  for (int im = 0; im < 2; ++im) {
#pragma unroll
    for (int in = 0; in < 2; ++in) {
      v8f acc = (im == 0) ? ((in == 0) ? acc00 : acc01)
                          : ((in == 0) ? acc10 : acc11);
      const int mB = mBase + im * 16;
      const int nB = nBase + in * 16;
      float vals[8];
#pragma unroll
      for (int r = 0; r < 8; ++r) {
        int m = mB + r + 8 * hi;
        float v = acc[r] + bias[m];
        if (do_mish) v = mishf(v);
        vals[r] = v;
      }
      if (outF) {
        size_t ob = (size_t)bz * M * N;
#pragma unroll
        for (int r = 0; r < 8; ++r) {
          size_t o = ob + (size_t)(mB + r + 8 * hi) * N + nB + l15;
          outF[o] = vals[r] + (addsrc ? addsrc[o] : 0.f);
        }
      }
      if (outT) {
        unsigned short* dst = outT + (size_t)bz * N * M + (size_t)(nB + l15) * M + mB + 8 * hi;
#pragma unroll
        for (int r = 0; r < 8; ++r) dst[r] = f2bf(vals[r]);
      }
    }
  }
}

// ---- cross-covariance attention per (b,head) -------------------------------
#define ACH 128
__global__ __launch_bounds__(256) void k_attn(const float* __restrict__ qkv,
                                              const float* __restrict__ temp,
                                              unsigned short* __restrict__ outT) {
  __shared__ float qs[DHEAD][ACH + 1];
  __shared__ float ks[DHEAD][ACH + 1];
  __shared__ float Ssm[DHEAD * DHEAD];
  __shared__ float invk[ACH];
  __shared__ float nqs[DHEAD];
  const int bh = blockIdx.x;
  const int b = bh >> 3, h = bh & 7;
  const float* q = qkv + ((size_t)b * (3 * ECH) + h * DHEAD) * TTOK;
  const float* k = q + (size_t)ECH * TTOK;
  const float* v = q + (size_t)(2 * ECH) * TTOK;
  const int tid = threadIdx.x;
  float accS[9];
#pragma unroll
  for (int j = 0; j < 9; ++j) accS[j] = 0.f;
  float nqacc = 0.f;
  for (int c = 0; c < TTOK / ACH; ++c) {
    const int t0 = c * ACH;
    for (int i = tid; i < DHEAD * ACH; i += 256) {
      int d = i / ACH, tc = i % ACH;
      qs[d][tc] = q[(size_t)d * TTOK + t0 + tc];
      ks[d][tc] = k[(size_t)d * TTOK + t0 + tc];
    }
    __syncthreads();
    if (tid < ACH) {
      float s = 0.f;
      for (int d = 0; d < DHEAD; ++d) { float kv = ks[d][tid]; s += kv * kv; }
      invk[tid] = 1.f / fmaxf(sqrtf(s), 1e-12f);
    }
    if (tid < DHEAD) {
      float s = 0.f;
      for (int tc = 0; tc < ACH; ++tc) { float qv = qs[tid][tc]; s += qv * qv; }
      nqacc += s;
    }
    __syncthreads();
#pragma unroll
    for (int j = 0; j < 9; ++j) {
      int idx = j * 256 + tid;           // 0..2303 == d*48+e
      int d = idx / DHEAD, e = idx % DHEAD;
      float a = 0.f;
      for (int tc = 0; tc < ACH; ++tc) a += qs[d][tc] * ks[e][tc] * invk[tc];
      accS[j] += a;
    }
    __syncthreads();
  }
#pragma unroll
  for (int j = 0; j < 9; ++j) Ssm[j * 256 + tid] = accS[j];
  if (tid < DHEAD) nqs[tid] = nqacc;
  __syncthreads();
  if (tid < DHEAD) {
    int d = tid;
    float scl = temp[h] / fmaxf(sqrtf(nqs[d]), 1e-12f);
    float mx = -1e30f;
    for (int e = 0; e < DHEAD; ++e) {
      float vv = Ssm[d * DHEAD + e] * scl;
      Ssm[d * DHEAD + e] = vv; mx = fmaxf(mx, vv);
    }
    float s = 0.f;
    for (int e = 0; e < DHEAD; ++e) {
      float p = __expf(Ssm[d * DHEAD + e] - mx);
      Ssm[d * DHEAD + e] = p; s += p;
    }
    float inv = 1.f / s;
    for (int e = 0; e < DHEAD; ++e) Ssm[d * DHEAD + e] *= inv;
  }
  __syncthreads();
  // out = attn @ v, write bf16 token-major
  for (int c = 0; c < TTOK / ACH; ++c) {
    const int t0 = c * ACH;
    for (int i = tid; i < DHEAD * ACH; i += 256) {
      int d = i / ACH, tc = i % ACH;
      ks[d][tc] = v[(size_t)d * TTOK + t0 + tc];
    }
    __syncthreads();
    for (int i = tid; i < DHEAD * ACH; i += 256) {
      int d = i / ACH, tc = i % ACH;
      float a = 0.f;
#pragma unroll 8
      for (int e = 0; e < DHEAD; ++e) a += Ssm[d * DHEAD + e] * ks[e][tc];
      outT[((size_t)b * TTOK + t0 + tc) * ECH + h * DHEAD + d] = f2bf(a);
    }
    __syncthreads();
  }
}

// ---- y = x + proj + dwconv3x3(xn) + rpe_b ----------------------------------
__global__ void k_resid(const float* __restrict__ x, const float* __restrict__ proj,
                        const unsigned short* __restrict__ xnT,
                        const float* __restrict__ rpw, const float* __restrict__ rpb,
                        float* __restrict__ y) {
  size_t idx = (size_t)blockIdx.x * 256 + threadIdx.x;
  int t = (int)(idx & (TTOK - 1));
  int e = (int)((idx >> 12) % ECH);
  int b = (int)(idx / ((size_t)ECH * TTOK));
  int hh = t >> 6, ww = t & 63;
  float acc = x[idx] + proj[idx] + rpb[e];
#pragma unroll
  for (int dy = -1; dy <= 1; ++dy) {
    int h2 = hh + dy;
    if (h2 < 0 || h2 >= HGT) continue;
#pragma unroll
    for (int dx = -1; dx <= 1; ++dx) {
      int w2 = ww + dx;
      if (w2 < 0 || w2 >= WID) continue;
      float xv = bf2f(xnT[((size_t)b * TTOK + (h2 << 6) + w2) * ECH + e]);
      acc += xv * rpw[e * 9 + (dy + 1) * 3 + (dx + 1)];
    }
  }
  y[idx] = acc;
}

extern "C" void kernel_launch(void* const* d_in, const int* in_sizes, int n_in,
                              void* d_out, int out_size, void* d_ws, size_t ws_size,
                              hipStream_t stream) {
  const float* x     = (const float*)d_in[0];
  const float* ln1w  = (const float*)d_in[1];
  const float* ln1b  = (const float*)d_in[2];
  const float* qw    = (const float*)d_in[3];
  const float* qb    = (const float*)d_in[4];
  const float* qg    = (const float*)d_in[5];
  const float* qbe   = (const float*)d_in[6];
  const float* qm    = (const float*)d_in[7];
  const float* qv_   = (const float*)d_in[8];
  const float* kw    = (const float*)d_in[9];
  const float* kb    = (const float*)d_in[10];
  const float* kg    = (const float*)d_in[11];
  const float* kbe   = (const float*)d_in[12];
  const float* km    = (const float*)d_in[13];
  const float* kv_   = (const float*)d_in[14];
  const float* vw    = (const float*)d_in[15];
  const float* vb    = (const float*)d_in[16];
  const float* vg    = (const float*)d_in[17];
  const float* vbe   = (const float*)d_in[18];
  const float* vm    = (const float*)d_in[19];
  const float* vv_   = (const float*)d_in[20];
  const float* temp  = (const float*)d_in[21];
  const float* rpw   = (const float*)d_in[22];
  const float* rpb   = (const float*)d_in[23];
  const float* pw    = (const float*)d_in[24];
  const float* pb    = (const float*)d_in[25];
  const float* ln2w  = (const float*)d_in[26];
  const float* ln2b  = (const float*)d_in[27];
  const float* f1w   = (const float*)d_in[28];
  const float* f1b   = (const float*)d_in[29];
  const float* f2w   = (const float*)d_in[30];
  const float* f2b   = (const float*)d_in[31];

  unsigned char* ws = (unsigned char*)d_ws;
  size_t off = 0;
  auto alloc = [&](size_t bytes) -> void* {
    void* p = ws + off;
    off = (off + bytes + 255) & ~(size_t)255;
    return p;
  };
  unsigned short* xn_t = (unsigned short*)alloc((size_t)BATCH * TTOK * ECH * 2);
  unsigned short* x2_t = (unsigned short*)alloc((size_t)BATCH * TTOK * ECH * 2);
  unsigned short* at_t = (unsigned short*)alloc((size_t)BATCH * TTOK * ECH * 2);
  unsigned char*  big  = (unsigned char*)alloc((size_t)BATCH * 3 * ECH * TTOK * 4);
  float*          qkv  = (float*)big;                                   // live: steps 3-4
  unsigned short* h_t  = (unsigned short*)big;                          // live: steps 8-9
  float* proj_tmp = (float*)(big + (size_t)BATCH * TTOK * HIDC * 2);    // live: steps 5-6
  float* yb = (float*)alloc((size_t)BATCH * ECH * TTOK * 4);
  unsigned short* wqkv  = (unsigned short*)alloc((size_t)3 * ECH * ECH * 2);
  float*          bqkv  = (float*)alloc((size_t)3 * ECH * 4);
  unsigned short* wproj = (unsigned short*)alloc((size_t)ECH * ECH * 2);
  unsigned short* wfc1  = (unsigned short*)alloc((size_t)HIDC * ECH * 2);
  unsigned short* wfc2  = (unsigned short*)alloc((size_t)ECH * HIDC * 2);

  // 1) weight prep (BN folded into qkv weights/bias; bf16 conversions)
  k_prep_qkvw<<<(3 * ECH * ECH + 255) / 256, 256, 0, stream>>>(qw, kw, vw, qg, qv_, kg, kv_, vg, vv_, wqkv);
  k_prep_qkvb<<<(3 * ECH + 255) / 256, 256, 0, stream>>>(qb, qg, qbe, qm, qv_, kb, kg, kbe, km, kv_,
                                                         vb, vg, vbe, vm, vv_, bqkv);
  k_f2bfk<<<(ECH * ECH + 255) / 256, 256, 0, stream>>>(pw, wproj, ECH * ECH);
  k_f2bfk<<<(HIDC * ECH + 255) / 256, 256, 0, stream>>>(f1w, wfc1, HIDC * ECH);
  k_f2bfk<<<(ECH * HIDC + 255) / 256, 256, 0, stream>>>(f2w, wfc2, ECH * HIDC);
  // 2) LN1 -> xn (bf16 token-major)
  k_cln<<<BATCH * TTOK / 8, 256, 0, stream>>>(x, ln1w, ln1b, xn_t);
  // 3) qkv = fused(Wqkv) * xn  (WMMA; block tile 64x128, wave tile 32x32)
  k_gemm<<<dim3(TTOK / 128, (3 * ECH) / 64, BATCH), 256, 0, stream>>>(
      wqkv, xn_t, bqkv, nullptr, qkv, nullptr, 3 * ECH, TTOK, ECH, 0);
  // 4) cross-covariance attention
  k_attn<<<BATCH * HEADS, 256, 0, stream>>>(qkv, temp, at_t);
  // 5) proj GEMM (WMMA)
  k_gemm<<<dim3(TTOK / 128, ECH / 64, BATCH), 256, 0, stream>>>(
      wproj, at_t, pb, nullptr, proj_tmp, nullptr, ECH, TTOK, ECH, 0);
  // 6) y = x + proj + dwconv3x3(xn)
  k_resid<<<(int)(((size_t)BATCH * ECH * TTOK) / 256), 256, 0, stream>>>(x, proj_tmp, xn_t, rpw, rpb, yb);
  // 7) LN2 -> x2 (bf16 token-major)
  k_cln<<<BATCH * TTOK / 8, 256, 0, stream>>>(yb, ln2w, ln2b, x2_t);
  // 8) h = mish(fc1(x2))  (WMMA, bf16 token-major out)
  k_gemm<<<dim3(TTOK / 128, HIDC / 64, BATCH), 256, 0, stream>>>(
      wfc1, x2_t, f1b, nullptr, nullptr, h_t, HIDC, TTOK, ECH, 1);
  // 9) out = y + fc2(h)  (WMMA, f32 out)
  k_gemm<<<dim3(TTOK / 128, ECH / 64, BATCH), 256, 0, stream>>>(
      wfc2, h_t, f2b, yb, (float*)d_out, nullptr, ECH, TTOK, HIDC, 0);
}
